// RNNDecoder_21698174779454
// MI455X (gfx1250) — matrix operand order repaired and checked by reading, not compiled, not run on hardware
//
#include <hip/hip_runtime.h>

// ---------------- problem constants ----------------
#define BB 32
#define TT 64
#define II 300
#define HH 168
#define CC 168
#define SS 512
#define VV 50257

#define KDEC 192            // K for decode GEMM: 168 padded to multiple of 32
#define KG   640            // K for gates GEMM: 300+168+168=636 padded to 640
#define NG   672            // 4*H gate outputs
#define NVPAD 50304         // V padded to 786*4*16

#define SAD  (KDEC + 8)     // LDS row stride (bf16) for decode A tile (anti-bank-conflict)
#define SAG  (KG + 8)       // LDS row stride (bf16) for gates A tile

typedef __attribute__((ext_vector_type(16))) __bf16 v16bf;
typedef __attribute__((ext_vector_type(8)))  __bf16 v8bf;
typedef __attribute__((ext_vector_type(8)))  float  v8f;

__device__ __forceinline__ v8f wmma_bf16(v16bf a, v16bf b, v8f c) {
    // (neg_a, A, neg_b, B, c_mod, C, reuse_a, reuse_b)
    return __builtin_amdgcn_wmma_f32_16x16x32_bf16(false, a, false, b, (short)0, c,
                                                   false, false);
}

__device__ __forceinline__ v16bf mk16(v8bf lo, v8bf hi) {
    v16bf r;
#pragma unroll
    for (int i = 0; i < 8; ++i) { r[i] = lo[i]; r[8 + i] = hi[i]; }
    return r;
}

// 16-bit A-matrix 16x32 fragment: lane<16 -> K0..15, lane>=16 -> K16..31
__device__ __forceinline__ v16bf load_a_frag(const __bf16* row, int kb, int lane) {
    const int ko = (lane < 16) ? 0 : 16;
    v8bf lo = *(const v8bf*)(row + kb + ko);
    v8bf hi = *(const v8bf*)(row + kb + ko + 8);
    return mk16(lo, hi);
}

// 16-bit B-matrix 32x16 fragment (contiguous k, same order as A): lane = N column, lane<16 -> K0..15, lane>=16 -> K16..31
__device__ __forceinline__ v16bf load_b_frag(const __bf16* row, int kb, int lane) {
    const int ko = (lane < 16) ? 0 : 16;
    return *(const v16bf*)(row + kb + ko);
}

__device__ __forceinline__ float sigf(float x) { return 1.0f / (1.0f + expf(-x)); }

// ---------------- prep kernels (run once per call; deterministic) ----------------

// W_dec (V x H f32) -> bf16, K padded to KDEC, N padded to NVPAD (zeros)
__global__ void k_conv_wdec(const float* __restrict__ wdec, __bf16* __restrict__ out) {
    long idx = (long)blockIdx.x * blockDim.x + threadIdx.x;
    const long tot = (long)NVPAD * KDEC;
    if (idx >= tot) return;
    int n = (int)(idx / KDEC), k = (int)(idx % KDEC);
    float v = (n < VV && k < HH) ? wdec[(long)n * HH + k] : 0.0f;
    out[idx] = (__bf16)v;
}

// fuse [W_ih (NGx468) | W_hh (NGx168)] -> bf16 NG x KG (cols 636..639 zero)
__global__ void k_conv_wgates(const float* __restrict__ wih, const float* __restrict__ whh,
                              __bf16* __restrict__ out) {
    int idx = blockIdx.x * blockDim.x + threadIdx.x;
    if (idx >= NG * KG) return;
    int n = idx / KG, k = idx % KG;
    float v = 0.0f;
    if (k < II + CC)            v = wih[n * (II + CC) + k];
    else if (k < II + CC + HH)  v = whh[n * HH + (k - II - CC)];
    out[idx] = (__bf16)v;
}

// zero c state, h_bf (incl. K padding), rnn_in
__global__ void k_init_state(float* __restrict__ c, __bf16* __restrict__ hbf,
                             __bf16* __restrict__ rnn) {
    int idx = blockIdx.x * blockDim.x + threadIdx.x;
    if (idx < BB * KG)  rnn[idx] = (__bf16)0.0f;
    if (idx < BB * KDEC) hbf[idx] = (__bf16)0.0f;
    if (idx < BB * HH)   c[idx] = 0.0f;
}

// seq_proj[b,s,a] = sum_c context[b,s,c] * att_Ws[c,a]   (time-invariant)
__global__ void k_seqproj(const float* __restrict__ context, const float* __restrict__ att_Ws,
                          float* __restrict__ seq_proj) {
    __shared__ float sc[CC];
    const int bs = blockIdx.x;            // b*SS + s
    const int a = threadIdx.x;
    for (int j = threadIdx.x; j < CC; j += blockDim.x) sc[j] = context[(long)bs * CC + j];
    __syncthreads();
    if (a < HH) {
        float acc = 0.0f;
        for (int j = 0; j < CC; ++j) acc += sc[j] * att_Ws[j * HH + a];
        seq_proj[(long)bs * HH + a] = acc;
    }
}

// ---------------- per-step kernel 1: additive attention ----------------
__global__ __launch_bounds__(256)
void k_attn(const float* __restrict__ x, const float* __restrict__ context,
            const int* __restrict__ ctx_len, const float* __restrict__ att_Wx,
            const float* __restrict__ att_b, const float* __restrict__ att_v,
            const float* __restrict__ seq_proj, const __bf16* __restrict__ hbf,
            __bf16* __restrict__ rnn, int t) {
    __shared__ float sh[HH], sv[HH], shwx[HH], se[SS], red[256];
    const int b = blockIdx.x, tid = threadIdx.x;

    for (int j = tid; j < HH; j += blockDim.x) {
        sh[j] = (float)hbf[b * KDEC + j];
        sv[j] = att_v[j];
    }
    __syncthreads();

    for (int a = tid; a < HH; a += blockDim.x) {
        float acc = att_b[a];
        for (int j = 0; j < HH; ++j) acc += sh[j] * att_Wx[j * HH + a];
        shwx[a] = acc;
    }
    __syncthreads();

    const int L = ctx_len[b];
    for (int s = tid; s < SS; s += blockDim.x) {
        const float* sp = seq_proj + ((long)(b * SS + s)) * HH;
        float acc = 0.0f;
        for (int a = 0; a < HH; ++a) acc += sv[a] * tanhf(shwx[a] + sp[a]);
        se[s] = (s < L) ? acc : -1e9f;
    }
    __syncthreads();

    // masked softmax over 512
    float m = fmaxf(se[tid], se[tid + 256]);
    red[tid] = m; __syncthreads();
    for (int w = 128; w > 0; w >>= 1) { if (tid < w) red[tid] = fmaxf(red[tid], red[tid + w]); __syncthreads(); }
    const float mx = red[0];
    __syncthreads();
    float lsum = 0.0f;
    for (int s = tid; s < SS; s += blockDim.x) { float e = expf(se[s] - mx); se[s] = e; lsum += e; }
    red[tid] = lsum; __syncthreads();
    for (int w = 128; w > 0; w >>= 1) { if (tid < w) red[tid] += red[tid + w]; __syncthreads(); }
    const float inv = 1.0f / red[0];
    __syncthreads();
    for (int s = tid; s < SS; s += blockDim.x) se[s] *= inv;
    __syncthreads();

    // attn_ctx and x_t into bf16 A-matrix rows
    for (int c = tid; c < CC; c += blockDim.x) {
        float acc = 0.0f;
        for (int s = 0; s < SS; ++s) acc += se[s] * context[((long)(b * SS + s)) * CC + c];
        rnn[b * KG + II + c] = (__bf16)acc;
    }
    for (int k = tid; k < II; k += blockDim.x)
        rnn[b * KG + k] = (__bf16)x[((long)b * TT + t) * II + k];
}

// ---------------- per-step kernel 2: gates GEMM (WMMA, A in LDS) + LSTM pointwise ----------------
__global__ __launch_bounds__(256)
void k_gates(const __bf16* __restrict__ rnn, const __bf16* __restrict__ wg,
             const float* __restrict__ b_ih, const float* __restrict__ b_hh,
             float* __restrict__ c_state, __bf16* __restrict__ hbf,
             __bf16* __restrict__ rnn_out) {
    extern __shared__ char smem[];
    float*  sgates = (float*)smem;                       // BB*NG f32 = 86016 B
    __bf16* sA     = (__bf16*)(smem + BB * NG * 4);      // BB x SAG bf16 = 41472 B
    const int tid = threadIdx.x;
    const int wave = tid >> 5, lane = tid & 31;

    // stage A (rnn_in) into LDS, padded row stride
    for (int i = tid; i < BB * (KG / 8); i += blockDim.x) {
        const int r = i / (KG / 8), cc = (i % (KG / 8)) * 8;
        *(v8bf*)(sA + r * SAG + cc) = *(const v8bf*)(rnn + r * KG + cc);
    }
    __syncthreads();

    for (int tile = wave; tile < 2 * (NG / 16); tile += 8) {
        const int mtile = tile & 1, ntile = tile >> 1;
        const __bf16* rowA = sA + (mtile * 16 + (lane & 15)) * SAG;
        const __bf16* rowB = wg + (ntile * 16 + (lane & 15)) * KG;
        v8f acc = {};
#pragma unroll 4
        for (int kc = 0; kc < KG / 32; ++kc) {
            const int kb = kc * 32;
            v16bf vb = load_b_frag(rowB, kb, lane);
            v16bf va = load_a_frag(rowA, kb, lane);
            acc = wmma_bf16(va, vb, acc);
        }
        const int nl = lane & 15;
        const int n = ntile * 16 + nl;
        const float bias = b_ih[n] + b_hh[n];
#pragma unroll
        for (int r = 0; r < 8; ++r) {
            const int ml = r + ((lane < 16) ? 0 : 8);
            sgates[(mtile * 16 + ml) * NG + n] = acc[r] + bias;
        }
    }
    __syncthreads();

    for (int idx = tid; idx < BB * HH; idx += blockDim.x) {
        const int b = idx / HH, j = idx % HH;
        const float* g = &sgates[b * NG];
        const float ig = sigf(g[j]);
        const float fg = sigf(g[HH + j]);
        const float gg = tanhf(g[2 * HH + j]);
        const float og = sigf(g[3 * HH + j]);
        const float cn = fg * c_state[idx] + ig * gg;
        c_state[idx] = cn;
        const float hn = og * tanhf(cn);
        const __bf16 hb = (__bf16)hn;
        hbf[b * KDEC + j] = hb;                 // decode A-matrix (K-padded, pad stays 0)
        rnn_out[b * KG + II + CC + j] = hb;     // recurrent columns for next step
    }
}

// ---------------- per-step kernel 3: decode GEMM (WMMA, A in LDS, B double-buffered) ----------------
__global__ __launch_bounds__(128)
void k_decode(const __bf16* __restrict__ hbf, const __bf16* __restrict__ wdec,
              float* __restrict__ out, int t) {
    __shared__ __bf16 sA[BB * SAD];             // 12800 B, shared by all 4 waves
    const int tid = threadIdx.x;
    const int wave = tid >> 5, lane = tid & 31;

    for (int i = tid; i < BB * (KDEC / 8); i += blockDim.x) {
        const int r = i / (KDEC / 8), cc = (i % (KDEC / 8)) * 8;
        *(v8bf*)(sA + r * SAD + cc) = *(const v8bf*)(hbf + r * KDEC + cc);
    }
    __syncthreads();

    const int nbase = (blockIdx.x * 4 + wave) * 16;
    const __bf16* rowB  = wdec + (long)(nbase + (lane & 15)) * KDEC;
    const __bf16* rowA0 = sA + (lane & 15) * SAD;
    const __bf16* rowA1 = sA + (16 + (lane & 15)) * SAD;

    v8f acc0 = {}, acc1 = {};
    v16bf vb = load_b_frag(rowB, 0, lane);      // prefetch first B fragment
#pragma unroll
    for (int kc = 0; kc < KDEC / 32; ++kc) {
        const int kb = kc * 32;
        v16bf vbn = vb;
        if (kc + 1 < KDEC / 32) vbn = load_b_frag(rowB, kb + 32, lane);  // prefetch next B
        v16bf va0 = load_a_frag(rowA0, kb, lane);                        // ds_load
        v16bf va1 = load_a_frag(rowA1, kb, lane);
        acc0 = wmma_bf16(va0, vb, acc0);
        acc1 = wmma_bf16(va1, vb, acc1);
        vb = vbn;
    }

    const int n = nbase + (lane & 15);
    if (n < VV) {
        const long tv = (long)TT * VV;
#pragma unroll
        for (int r = 0; r < 8; ++r) {
            const int ml = r + ((lane < 16) ? 0 : 8);
            out[(long)ml * tv + (long)t * VV + n] = acc0[r];
            out[(long)(16 + ml) * tv + (long)t * VV + n] = acc1[r];
        }
    }
}

// ---------------- host launch ----------------
extern "C" void kernel_launch(void* const* d_in, const int* in_sizes, int n_in,
                              void* d_out, int out_size, void* d_ws, size_t ws_size,
                              hipStream_t stream) {
    const float* x        = (const float*)d_in[0];
    const float* context  = (const float*)d_in[1];
    const int*   ctx_len  = (const int*)d_in[2];
    const float* W_ih     = (const float*)d_in[3];
    const float* W_hh     = (const float*)d_in[4];
    const float* b_ih     = (const float*)d_in[5];
    const float* b_hh     = (const float*)d_in[6];
    const float* att_Wx   = (const float*)d_in[7];
    const float* att_Ws   = (const float*)d_in[8];
    const float* att_b    = (const float*)d_in[9];
    const float* att_v    = (const float*)d_in[10];
    const float* W_dec    = (const float*)d_in[11];
    float* out = (float*)d_out;
    char* ws = (char*)d_ws;

    // workspace layout (all offsets 256B-aligned)
    size_t off = 0;
    __bf16* wdec_bf = (__bf16*)(ws + off); off += (size_t)NVPAD * KDEC * 2;   // 19.3 MB
    __bf16* wg_bf   = (__bf16*)(ws + off); off += (size_t)NG * KG * 2;        // 0.86 MB
    float*  seqproj = (float*)(ws + off);  off += (size_t)BB * SS * HH * 4;   // 11.0 MB
    __bf16* rnn_in  = (__bf16*)(ws + off); off += (size_t)BB * KG * 2;
    __bf16* h_bf    = (__bf16*)(ws + off); off += (size_t)BB * KDEC * 2;
    float*  c_state = (float*)(ws + off);  off += (size_t)BB * HH * 4;
    (void)ws_size; (void)in_sizes; (void)n_in; (void)out_size;

    // --- prep (once per call) ---
    {
        long tot = (long)NVPAD * KDEC;
        k_conv_wdec<<<(unsigned)((tot + 255) / 256), 256, 0, stream>>>(W_dec, wdec_bf);
        k_conv_wgates<<<(NG * KG + 255) / 256, 256, 0, stream>>>(W_ih, W_hh, wg_bf);
        k_init_state<<<(BB * KG + 255) / 256, 256, 0, stream>>>(c_state, h_bf, rnn_in);
        k_seqproj<<<BB * SS, 192, 0, stream>>>(context, att_Ws, seqproj);
    }

    const unsigned gates_lds = BB * NG * sizeof(float) + BB * SAG * sizeof(__bf16);

    // --- recurrence over T ---
    for (int t = 0; t < TT; ++t) {
        k_attn<<<BB, 256, 0, stream>>>(x, context, ctx_len, att_Wx, att_b, att_v,
                                       seqproj, h_bf, rnn_in, t);
        k_gates<<<1, 256, gates_lds, stream>>>(rnn_in, wg_bf, b_ih, b_hh,
                                               c_state, h_bf, rnn_in);
        k_decode<<<NVPAD / 64, 128, 0, stream>>>(h_bf, wdec_bf, out, t);
    }
}
